// LatentSDE_44117904064573
// MI455X (gfx1250) — matrix operand order repaired and checked by reading, not compiled, not run on hardware
//
#include <hip/hip_runtime.h>
#include <hip/hip_bf16.h>

#define E_ 7
#define B_ 50
#define T_ 100
#define D_ 128
#define A_ 32
#define L_ 256
#define H_ 1024
#define C_ 128
#define LAD_ (L_ + A_ + D_)   // 416
#define EB_ (E_ * B_)         // 350
#define H_STEP_F 0.1f
#define SQRT_H_F 0.31622776601683794f

typedef __bf16 bf16;
typedef __attribute__((ext_vector_type(2)))  __bf16 v2bf;
typedef __attribute__((ext_vector_type(16))) __bf16 v16bf;
typedef __attribute__((ext_vector_type(8)))  float  v8f;

__device__ __forceinline__ unsigned pack2bf(float x, float y) {
  v2bf v;
  v[0] = (bf16)x;
  v[1] = (bf16)y;
  return __builtin_bit_cast(unsigned, v);   // -> v_cvt_pk_bf16_f32
}

// ---------------------------------------------------------------------------
// Batched GEMM:  Y[z] = act( X[z] (MxK) @ W[z] (KxN) + bias[z] )
// fp32 in/out, v_wmma_f32_16x16x32_bf16 with fp32 accumulation.
// N is a compile-time parameter so all B-tile loads use one base address
// plus immediate offsets (k*N*4 <= 254KB < 2^23) — no per-load address
// arithmetic / xcnt waits in the staging phase.
// grid: (ceil(M/16), N/128, batch); block: 128 threads (4 wave32s).
// Each wave computes two 16x16 N-subtiles (2 WMMA / K-step, shared A frag).
// LDS tiles staged directly in WMMA fragment layout (lane's operand = one
// contiguous 32B chunk -> ds_load_b128 x2).  Double-buffered K loop with
// software-pipelined staging; out-of-range M rows are CLAMPED (their outputs
// are masked at the store).  Requires N%128==0, K%32==0.
// ACT: 0=none, 1=relu, 2=tanh
// ---------------------------------------------------------------------------
template <int ACT, int N>
__global__ __launch_bounds__(128) void gemm_bias_act(
    const float* __restrict__ X, long long sxb,
    const float* __restrict__ W, long long swb,
    const float* __restrict__ Bias, long long sbb,
    float* __restrict__ Y, long long syb,
    int M, int K)
{
  // [buf][lane][elem] : per-lane fragment storage (32B per lane, aligned)
  __shared__ __align__(32) bf16 Af[2][32][16];        // A 16x32 tile
  __shared__ __align__(32) bf16 Bf[2][8][32][16];     // B 32x128 tile (8 subtiles)

  const float* Xb = X + (long long)blockIdx.z * sxb;
  const float* Wb = W + (long long)blockIdx.z * swb;
  const float* Bb = Bias + (long long)blockIdx.z * sbb;
  float*       Yb = Y + (long long)blockIdx.z * syb;

  const int m0   = blockIdx.x * 16;
  const int n0   = blockIdx.y * 128;
  const int lane = threadIdx.x & 31;
  const int wave = threadIdx.x >> 5;

  // Per-thread staging coordinates (invariant over K steps)
  const int ar  = threadIdx.x >> 4;                // A row (it=0); +8 for it=1
  const int ac  = (threadIdx.x & 15) * 2;          // A even-K col
  const int adl = 16 * ((ac & 15) >> 3);           // + row
  const int adj = ((ac >> 4) << 3) + (ac & 7);
  const int bn  = threadIdx.x & 127;

  // Clamped A row indices: rows >= M read row M-1 (garbage only feeds
  // output rows >= M, which are masked at the epilogue store).
  int arow[2];
#pragma unroll
  for (int it = 0; it < 2; ++it) {
    int r = m0 + ar + it * 8;
    arow[it] = (r < M) ? r : (M - 1);
  }

  v8f acc0 = {};
  v8f acc1 = {};

  // Stage one K-step (32-wide) of A and B into fragment-layout LDS.
  // Phase 1 issues every global load; phase 2 converts + stores to LDS.
  auto load_tiles = [&](int k0, int buf) {
    float av0[2], av1[2];
#pragma unroll
    for (int it = 0; it < 2; ++it) {
      const float2 f2 =
          *(const float2*)(Xb + (long long)arow[it] * K + (k0 + ac));
      av0[it] = f2.x; av1[it] = f2.y;
    }
    // Single base pointer; all 32 offsets are compile-time immediates.
    const float* bbase = Wb + (long long)k0 * N + (n0 + bn);
    float bv0[16], bv1[16];
#pragma unroll
    for (int it = 0; it < 16; ++it) {
      bv0[it] = bbase[(2 * it) * N];
      bv1[it] = bbase[(2 * it + 1) * N];
    }
#pragma unroll
    for (int it = 0; it < 2; ++it) {
      const int r = ar + it * 8;
      *(unsigned*)&Af[buf][r + adl][adj] = pack2bf(av0[it], av1[it]);
    }
#pragma unroll
    for (int it = 0; it < 16; ++it) {
      const int k = it * 2;
      const int dl = (bn & 15) + 16 * ((k >> 3) & 1);
      *(unsigned*)&Bf[buf][bn >> 4][dl][((k >> 4) << 3) + (k & 7)] = pack2bf(bv0[it], bv1[it]);
    }
  };

  load_tiles(0, 0);
  __syncthreads();

  int buf = 0;
  for (int k0 = 0; k0 < K; k0 += 32) {
    v16bf a  = *(const v16bf*)&Af[buf][lane][0];
    v16bf b0 = *(const v16bf*)&Bf[buf][2 * wave][lane][0];
    v16bf b1 = *(const v16bf*)&Bf[buf][2 * wave + 1][lane][0];
    if (k0 + 32 < K) load_tiles(k0 + 32, buf ^ 1);
    acc0 = __builtin_amdgcn_wmma_f32_16x16x32_bf16(
        false, a, false, b0, (short)0, acc0, false, false);
    acc1 = __builtin_amdgcn_wmma_f32_16x16x32_bf16(
        false, a, false, b1, (short)0, acc1, false, false);
    __syncthreads();
    buf ^= 1;
  }

  // C/D layout: VGPR r -> (M = r + 8*(lane>>4), N = lane&15)
  const int   row   = lane & 15;
  const int   hi    = lane >> 4;
  const int   cn0   = n0 + (2 * wave) * 16 + row;
  const int   cn1   = cn0 + 16;
  const float bias0 = Bb[cn0];
  const float bias1 = Bb[cn1];
#pragma unroll
  for (int r = 0; r < 8; ++r) {
    int cm = m0 + 8 * hi + r;
    if (cm < M) {
      float v0 = acc0[r] + bias0;
      float v1 = acc1[r] + bias1;
      if (ACT == 1) { v0 = fmaxf(v0, 0.0f); v1 = fmaxf(v1, 0.0f); }
      if (ACT == 2) { v0 = tanhf(v0);       v1 = tanhf(v1); }
      Yb[(long long)cm * N + cn0] = v0;
      Yb[(long long)cm * N + cn1] = v1;
    }
  }
}

// ---------------------------------------------------------------------------
// z0 + KL from q = [qm | ql] (E, rowsPerE, 2L).  One block per (row, e).
// ---------------------------------------------------------------------------
__global__ __launch_bounds__(256) void z0_kl_kernel(
    const float* __restrict__ q, int rowsPerE,
    const float* __restrict__ eps_z0,
    const float* __restrict__ pm, const float* __restrict__ pl,
    float* __restrict__ z_state, float* __restrict__ kl_rows, int t0)
{
  const int r = blockIdx.x;
  const int e = blockIdx.y;
  const int l = threadIdx.x;
  const int t = t0 + r / B_;
  const int b = r % B_;
  const float* qrow = q + ((long long)e * rowsPerE + r) * (2 * L_);
  float qm  = qrow[l];
  float ql  = qrow[L_ + l];
  float pmv = pm[e * L_ + l];
  float plv = pl[e * L_ + l];
  float d   = qm - pmv;
  float kl  = plv - ql + (expf(2.0f * ql) + d * d) / (2.0f * expf(2.0f * plv)) - 0.5f;
  if (t == 0) {
    float ev = eps_z0[(((long long)0 * E_ + e) * B_ + b) * L_ + l];
    z_state[((long long)e * B_ + b) * L_ + l] = qm + expf(ql) * ev;
  }
  __shared__ float red[L_];
  red[l] = kl;
  __syncthreads();
  for (int s = L_ / 2; s > 0; s >>= 1) {
    if (l < s) red[l] += red[l + s];
    __syncthreads();
  }
  if (l == 0) kl_rows[(long long)e * (T_ * B_) + t * B_ + b] = red[0];
}

// lad = concat([z_state, actions[:,t], xs[:,t]], axis=-1) : (E,B,416)
__global__ void build_lad_kernel(
    const float* __restrict__ z_state, const float* __restrict__ actions,
    const float* __restrict__ xs, float* __restrict__ lad, int t)
{
  const int eb = blockIdx.x;
  const int c  = threadIdx.x;      // 0..415
  const int e  = eb / B_;
  const int b  = eb % B_;
  float v;
  if (c < L_)
    v = z_state[(long long)eb * L_ + c];
  else if (c < L_ + A_)
    v = actions[((long long)e * T_ * B_ + (long long)t * B_ + b) * A_ + (c - L_)];
  else
    v = xs[((long long)e * T_ * B_ + (long long)t * B_ + b) * D_ + (c - L_ - A_)];
  lad[(long long)eb * LAD_ + c] = v;
}

// g_fn + SDE update, one block per row eb, one thread per latent dim l.
__global__ __launch_bounds__(256) void g_update_kernel(
    const float* __restrict__ z, const float* __restrict__ fv,
    const float* __restrict__ hv,
    const float* __restrict__ gw1, const float* __restrict__ gb1,
    const float* __restrict__ gw2, const float* __restrict__ gb2,
    const float* __restrict__ dw, float* __restrict__ z_state,
    float* __restrict__ zs_all, float* __restrict__ lr_acc, int t)
{
  const int   eb = blockIdx.x;
  const int   l  = threadIdx.x;
  const float zv = z[(long long)eb * L_ + l];
  const float* w1 = gw1 + (long long)l * H_;
  const float* b1 = gb1 + (long long)l * H_;
  const float* w2 = gw2 + (long long)l * H_;
  float s = 0.0f;
#pragma unroll 4
  for (int hh = 0; hh < H_; ++hh) {
    float a = fmaxf(zv * w1[hh] + b1[hh], 0.0f);
    s += a * w2[hh];
  }
  float gv  = s + gb2[l];
  float fvl = fv[(long long)eb * L_ + l];
  float hvl = hv[(long long)eb * L_ + l];
  float u   = (fvl - hvl) / gv;
  __shared__ float red[L_];
  red[l] = u * u;
  __syncthreads();
  for (int st = L_ / 2; st > 0; st >>= 1) {
    if (l < st) red[l] += red[l + st];
    __syncthreads();
  }
  if (l == 0) lr_acc[eb] += 0.5f * red[0] * H_STEP_F;   // single writer per eb
  float zn = zv + fvl * H_STEP_F + gv * (SQRT_H_F * dw[(long long)eb * L_ + l]);
  z_state[(long long)eb * L_ + l] = zn;
  const int e = eb / B_;
  const int b = eb % B_;
  zs_all[(((long long)e * T_ + t) * B_ + b) * L_ + l] = zn;
}

// logqp_path[e] = (sum_{t,b} kl_rows + sum_b lr_acc) / B
__global__ __launch_bounds__(256) void finalize_kernel(
    const float* __restrict__ kl_rows, const float* __restrict__ lr_acc,
    float* __restrict__ out)
{
  const int e   = blockIdx.x;
  const int tid = threadIdx.x;
  float s = 0.0f;
  for (int i = tid; i < T_ * B_; i += 256) s += kl_rows[(long long)e * T_ * B_ + i];
  for (int i = tid; i < B_; i += 256) s += lr_acc[e * B_ + i];
  __shared__ float red[256];
  red[tid] = s;
  __syncthreads();
  for (int st = 128; st > 0; st >>= 1) {
    if (tid < st) red[tid] += red[tid + st];
    __syncthreads();
  }
  if (tid == 0) out[e] = red[0] / (float)B_;
}

static inline int ceil16(int x) { return (x + 15) / 16; }

extern "C" void kernel_launch(void* const* d_in, const int* in_sizes, int n_in,
                              void* d_out, int out_size, void* d_ws, size_t ws_size,
                              hipStream_t stream)
{
  (void)in_sizes; (void)n_in; (void)out_size; (void)ws_size;
  const float* enc_w1 = (const float*)d_in[0];
  const float* enc_b1 = (const float*)d_in[1];
  const float* enc_w2 = (const float*)d_in[2];
  const float* enc_b2 = (const float*)d_in[3];
  const float* enc_w3 = (const float*)d_in[4];
  const float* enc_b3 = (const float*)d_in[5];
  const float* qz0_w  = (const float*)d_in[6];
  const float* qz0_b  = (const float*)d_in[7];
  const float* act_w  = (const float*)d_in[8];
  const float* act_b  = (const float*)d_in[9];
  const float* proj_w1 = (const float*)d_in[10];
  const float* proj_b1 = (const float*)d_in[11];
  const float* proj_w2 = (const float*)d_in[12];
  const float* proj_b2 = (const float*)d_in[13];
  const float* proj_w3 = (const float*)d_in[14];
  const float* proj_b3 = (const float*)d_in[15];
  const float* f_w1 = (const float*)d_in[16];
  const float* f_b1 = (const float*)d_in[17];
  const float* f_w2 = (const float*)d_in[18];
  const float* f_b2 = (const float*)d_in[19];
  const float* f_w3 = (const float*)d_in[20];
  const float* f_b3 = (const float*)d_in[21];
  const float* h_w1 = (const float*)d_in[22];
  const float* h_b1 = (const float*)d_in[23];
  const float* h_w2 = (const float*)d_in[24];
  const float* h_b2 = (const float*)d_in[25];
  const float* h_w3 = (const float*)d_in[26];
  const float* h_b3 = (const float*)d_in[27];
  const float* g_w1 = (const float*)d_in[28];
  const float* g_b1 = (const float*)d_in[29];
  const float* g_w2 = (const float*)d_in[30];
  const float* g_b2 = (const float*)d_in[31];
  const float* pz0_mean   = (const float*)d_in[32];
  const float* pz0_logstd = (const float*)d_in[33];
  const float* xs      = (const float*)d_in[34];
  const float* actions = (const float*)d_in[35];
  const float* eps_z0  = (const float*)d_in[36];
  const float* dw_noise = (const float*)d_in[37];

  float* out = (float*)d_out;           // [0..6] logqp_path, [7..] predicted_xs
  float* ws  = (float*)d_ws;

  const int TC = 25;                    // timestep chunk for parallel phases
  const int RC = TC * B_;               // 1250 rows per ensemble member / chunk

  long long off = 0;
  float* bufA    = ws + off; off += (long long)E_ * RC * H_;
  float* bufB    = ws + off; off += (long long)E_ * RC * H_;
  float* bufQ    = ws + off; off += (long long)E_ * RC * 2 * L_;
  float* zs_all  = ws + off; off += (long long)E_ * T_ * B_ * L_;
  float* z_state = ws + off; off += (long long)EB_ * L_;
  float* lad     = ws + off; off += (long long)EB_ * LAD_;
  float* z_cur   = ws + off; off += (long long)EB_ * L_;
  float* f1b     = ws + off; off += (long long)EB_ * H_;
  float* h1b     = ws + off; off += (long long)EB_ * H_;
  float* f2b     = ws + off; off += (long long)EB_ * H_;
  float* h2b     = ws + off; off += (long long)EB_ * H_;
  float* fvb     = ws + off; off += (long long)EB_ * L_;
  float* hvb     = ws + off; off += (long long)EB_ * L_;
  float* kl_rows = ws + off; off += (long long)E_ * T_ * B_;
  float* lr_acc  = ws + off; off += (long long)EB_;

  hipMemsetAsync(lr_acc, 0, EB_ * sizeof(float), stream);

  const dim3 blk(128);

  // ---------------- Phase 1: encoder + qz0 + z0/KL (parallel over T) -------
  for (int t0 = 0; t0 < T_; t0 += TC) {
    const float* xchunk = xs + (long long)t0 * B_ * D_;
    gemm_bias_act<1, H_><<<dim3(ceil16(RC), H_ / 128, E_), blk, 0, stream>>>(
        xchunk, (long long)T_ * B_ * D_, enc_w1, (long long)D_ * H_,
        enc_b1, H_, bufA, (long long)RC * H_, RC, D_);
    gemm_bias_act<1, H_><<<dim3(ceil16(RC), H_ / 128, E_), blk, 0, stream>>>(
        bufA, (long long)RC * H_, enc_w2, (long long)H_ * H_,
        enc_b2, H_, bufB, (long long)RC * H_, RC, H_);
    gemm_bias_act<0, C_><<<dim3(ceil16(RC), C_ / 128, E_), blk, 0, stream>>>(
        bufB, (long long)RC * H_, enc_w3, (long long)H_ * C_,
        enc_b3, C_, bufA, (long long)RC * C_, RC, H_);
    gemm_bias_act<0, 2 * L_><<<dim3(ceil16(RC), (2 * L_) / 128, E_), blk, 0, stream>>>(
        bufA, (long long)RC * C_, qz0_w, (long long)C_ * 2 * L_,
        qz0_b, 2 * L_, bufQ, (long long)RC * 2 * L_, RC, C_);
    z0_kl_kernel<<<dim3(RC, E_), 256, 0, stream>>>(
        bufQ, RC, eps_z0, pz0_mean, pz0_logstd, z_state, kl_rows, t0);
  }

  // ---------------- Phase 2: sequential SDE integration --------------------
  for (int t = 0; t < T_; ++t) {
    build_lad_kernel<<<EB_, LAD_, 0, stream>>>(z_state, actions, xs, lad, t);
    gemm_bias_act<0, L_><<<dim3(ceil16(B_), L_ / 128, E_), blk, 0, stream>>>(
        lad, (long long)B_ * LAD_, act_w, (long long)LAD_ * L_,
        act_b, L_, z_cur, (long long)B_ * L_, B_, LAD_);
    gemm_bias_act<1, H_><<<dim3(ceil16(EB_), H_ / 128, 1), blk, 0, stream>>>(
        z_cur, 0, f_w1, 0, f_b1, 0, f1b, 0, EB_, L_);
    gemm_bias_act<1, H_><<<dim3(ceil16(EB_), H_ / 128, 1), blk, 0, stream>>>(
        z_cur, 0, h_w1, 0, h_b1, 0, h1b, 0, EB_, L_);
    gemm_bias_act<2, H_><<<dim3(ceil16(EB_), H_ / 128, 1), blk, 0, stream>>>(
        f1b, 0, f_w2, 0, f_b2, 0, f2b, 0, EB_, H_);
    gemm_bias_act<2, H_><<<dim3(ceil16(EB_), H_ / 128, 1), blk, 0, stream>>>(
        h1b, 0, h_w2, 0, h_b2, 0, h2b, 0, EB_, H_);
    gemm_bias_act<0, L_><<<dim3(ceil16(EB_), L_ / 128, 1), blk, 0, stream>>>(
        f2b, 0, f_w3, 0, f_b3, 0, fvb, 0, EB_, H_);
    gemm_bias_act<0, L_><<<dim3(ceil16(EB_), L_ / 128, 1), blk, 0, stream>>>(
        h2b, 0, h_w3, 0, h_b3, 0, hvb, 0, EB_, H_);
    g_update_kernel<<<EB_, L_, 0, stream>>>(
        z_cur, fvb, hvb, g_w1, g_b1, g_w2, g_b2,
        dw_noise + (long long)t * EB_ * L_, z_state, zs_all, lr_acc, t);
  }

  // ---------------- Phase 3: projector on all zs (parallel over T) ---------
  for (int t0 = 0; t0 < T_; t0 += TC) {
    const float* zchunk = zs_all + (long long)t0 * B_ * L_;
    gemm_bias_act<2, H_><<<dim3(ceil16(RC), H_ / 128, E_), blk, 0, stream>>>(
        zchunk, (long long)T_ * B_ * L_, proj_w1, (long long)L_ * H_,
        proj_b1, H_, bufA, (long long)RC * H_, RC, L_);
    gemm_bias_act<2, H_><<<dim3(ceil16(RC), H_ / 128, E_), blk, 0, stream>>>(
        bufA, (long long)RC * H_, proj_w2, (long long)H_ * H_,
        proj_b2, H_, bufB, (long long)RC * H_, RC, H_);
    gemm_bias_act<0, D_><<<dim3(ceil16(RC), D_ / 128, E_), blk, 0, stream>>>(
        bufB, (long long)RC * H_, proj_w3, (long long)H_ * D_,
        proj_b3, D_, out + 7 + (long long)t0 * B_ * D_,
        (long long)T_ * B_ * D_, RC, H_);
  }

  finalize_kernel<<<E_, 256, 0, stream>>>(kl_rows, lr_acc, out);
}
